// EuclideanTransformer_53154515255877
// MI455X (gfx1250) — matrix operand order repaired and checked by reading, not compiled, not run on hardware
//
#include <hip/hip_runtime.h>

// ---------------------------------------------------------------------------
// CDNA5 / gfx1250 implementation.
//   - All GEMMs via v_wmma_f32_16x16x32_f16 (f16 inputs, f32 accum).
//   - Weights pre-packed once into B-tile lane layout (f16) in d_ws.
//   - Activation relayout (C/D layout -> A layout) through per-wave LDS tiles.
//   - SiLU via gfx1250 v_tanh_f32: x*sigmoid(x) = t + t*tanh(t), t = x/2.
// ---------------------------------------------------------------------------

typedef __attribute__((ext_vector_type(16))) _Float16 v16h;
typedef __attribute__((ext_vector_type(8)))  _Float16 v8h;
typedef __attribute__((ext_vector_type(8)))  float    v8f;

#define N_NODES 50000
#define N_EDGES 800000
#define FDIM    128
#define NCOMP   16

union H16 { v16h v; v8h h[2]; };

__device__ __forceinline__ v8f wmma_f16(v16h a, v16h b, v8f c) {
  // (neg_a, A, neg_b, B, c_mod, C, reuse_a, reuse_b)
  return __builtin_amdgcn_wmma_f32_16x16x32_f16(false, a, false, b, (short)0, c,
                                                false, false);
}

// Packed B tile: [kc][nt][lane][16 halves]; lane n<16 -> col n, K=kc*32+j (j<8), kc*32+16+(j-8) (j>=8);
// lane n>=16 -> col n-16, K=kc*32+8+j (j<8), kc*32+24+(j-8) (j>=8).  (matches A tile K order)
__device__ __forceinline__ v16h load_btile(const _Float16* pk, int nNt, int kc,
                                           int nt, int lane) {
  const _Float16* t = pk + ((size_t)(kc * nNt + nt)) * 512 + lane * 16;
  H16 u;
  u.h[0] = *(const v8h*)t;
  u.h[1] = *(const v8h*)(t + 8);
  return u.v;
}

// A tile from a row-major 16 x W f16 LDS tile.
// lane<16: row=lane,   elems 0..7 = K kc*32+0..7,  elems 8..15 = K kc*32+16..23
// lane>=16: row=lane-16, elems 0..7 = K kc*32+8..15, 8..15 = K kc*32+24..31
__device__ __forceinline__ v16h load_atile(const _Float16* stage, int W,
                                           int lane, int kc) {
  const int row = lane & 15;
  const int off = row * W + kc * 32 + ((lane < 16) ? 0 : 8);
  H16 u;
  u.h[0] = *(const v8h*)(stage + off);
  u.h[1] = *(const v8h*)(stage + off + 16);
  return u.v;
}

// Store C/D-layout accumulator as f16 into row-major 16 x W LDS tile.
// lane<16: col=lane, rows 0..7; lane>=16: col=lane-16, rows 8..15.
__device__ __forceinline__ void store_dtile(_Float16* stage, int W, int lane,
                                            int nt, v8f acc) {
  const int col = (lane & 15) + nt * 16;
  const int r0 = (lane < 16) ? 0 : 8;
#pragma unroll
  for (int r = 0; r < 8; r++) stage[(r0 + r) * W + col] = (_Float16)acc[r];
}

__device__ __forceinline__ v8f bias_splat(const float* b, int nt, int lane) {
  const float v = b[nt * 16 + (lane & 15)];
  v8f a;
#pragma unroll
  for (int j = 0; j < 8; j++) a[j] = v;
  return a;
}

// silu(x) = x*sigmoid(x) = t + t*tanh(t), t = x/2.  1 TRANS + 1 mul + 1 fma.
__device__ __forceinline__ v8f silu8(v8f x) {
  v8f y;
#pragma unroll
  for (int j = 0; j < 8; j++) {
    const float t = 0.5f * x[j];
    y[j] = __builtin_fmaf(t, __builtin_amdgcn_tanhf(t), t);
  }
  return y;
}

// ---------------------------------------------------------------------------
// Weight pre-pack: f32 (K x N) row-major -> f16 B-tile layout, zero padded.
// grid.x = ceil(K/32)*ceil(N/16), block = 32.
// ---------------------------------------------------------------------------
__global__ void pack_weight_kernel(const float* __restrict__ W,
                                   _Float16* __restrict__ dst, int K, int N) {
  const int nNt = (N + 15) >> 4;
  const int tilei = blockIdx.x;
  const int kc = tilei / nNt, nt = tilei % nNt;
  const int lane = threadIdx.x;
  const int col = nt * 16 + (lane & 15);
  const int kb = kc * 32 + ((lane < 16) ? 0 : 8);
  _Float16* out = dst + (size_t)tilei * 512 + lane * 16;
#pragma unroll
  for (int j = 0; j < 16; j++) {
    const int k = kb + j + ((j >= 8) ? 8 : 0);
    const float v = (k < K && col < N) ? W[(size_t)k * N + col] : 0.f;
    out[j] = (_Float16)v;
  }
}

// ---------------------------------------------------------------------------
// Edge filter kernel: one wave per 16-edge tile, both filters (fi, fe).
// ---------------------------------------------------------------------------
struct EdgeParams {
  const float* lengths;
  const int* senders;
  const int* receivers;
  const float* ev;
  const _Float16* w1[2];
  const _Float16* w2[2];
  const _Float16* ew1[2];
  const _Float16* ew2[2];
  const float* br1[2];
  const float* br2[2];
  const float* be1[2];
  const float* be2[2];
  float* out[2];
};

__global__ __launch_bounds__(256) void edge_filter_kernel(EdgeParams p) {
  __shared__ _Float16 smem[8 * 16 * 128];
  const int lane = threadIdx.x & 31;
  const int wid = threadIdx.x >> 5;
  const long tile = (long)blockIdx.x * 8 + wid;
  _Float16* stage = smem + wid * (16 * 128);
  const int row = lane & 15;
  const long e0 = tile * 16;

  // L2 prefetch of the hot 128x128 weights.
  __builtin_prefetch(p.w2[0], 0, 3);
  __builtin_prefetch(p.w2[1], 0, 3);

  // ---- RBF A tile (16 x 32), built directly in A-layout registers --------
  const float d = p.lengths[e0 + row];
  const float GAMMA = 0.5f / ((5.0f / 31.0f) * (5.0f / 31.0f));
  v16h a_rbf;
  {
    const int kb = (lane >> 4) << 3;
#pragma unroll
    for (int j = 0; j < 16; j++) {
      const int k = kb + j + ((j >= 8) ? 8 : 0);
      const float c = (5.0f / 31.0f) * (float)k;
      const float t = d - c;
      a_rbf[j] = (_Float16)__expf(-GAMMA * t * t);
    }
  }

  // ---- SO(3) invariant A tile (16 x 32, only K=0..3 nonzero) -------------
  v16h a_ev;
#pragma unroll
  for (int j = 0; j < 16; j++) a_ev[j] = (_Float16)0.f;
  if (lane < 16) {
    const int s = p.senders[e0 + row];
    const int r = p.receivers[e0 + row];
    const float* es = p.ev + (size_t)s * NCOMP;
    const float* er = p.ev + (size_t)r * NCOMP;
    float i0 = 0.f, i1 = 0.f, i2 = 0.f, i3 = 0.f;
#pragma unroll
    for (int q = 0; q < NCOMP; q++) {
      const float dv = es[q] - er[q];
      const float sq = dv * dv;
      if (q == 0) i0 += sq;
      else if (q < 4) i1 += sq;
      else if (q < 9) i2 += sq;
      else i3 += sq;
    }
    a_ev[0] = (_Float16)i0;
    a_ev[1] = (_Float16)i1;
    a_ev[2] = (_Float16)i2;
    a_ev[3] = (_Float16)i3;
  }

#pragma unroll
  for (int f = 0; f < 2; f++) {
    // e-branch layer 1: (16x32 sparse) @ (32x32)
#pragma unroll
    for (int nt = 0; nt < 2; nt++) {
      v8f acc = bias_splat(p.be1[f], nt, lane);
      acc = wmma_f16(a_ev, load_btile(p.ew1[f], 2, 0, nt, lane), acc);
      store_dtile(stage, 32, lane, nt, silu8(acc));
    }
    __builtin_amdgcn_wave_barrier();
    const v16h a_e = load_atile(stage, 32, lane, 0);
    __builtin_amdgcn_wave_barrier();

    // h-branch layer 1: (16x32) @ (32x128)
#pragma unroll
    for (int nt = 0; nt < 8; nt++) {
      v8f acc = bias_splat(p.br1[f], nt, lane);
      acc = wmma_f16(a_rbf, load_btile(p.w1[f], 8, 0, nt, lane), acc);
      store_dtile(stage, 128, lane, nt, silu8(acc));
    }
    __builtin_amdgcn_wave_barrier();
    v16h a_h[4];
#pragma unroll
    for (int kc = 0; kc < 4; kc++) a_h[kc] = load_atile(stage, 128, lane, kc);
    __builtin_amdgcn_wave_barrier();

    // out = silu(h @ w2 + b2) + silu(e @ ew2 + eb2)
    const int r0 = (lane < 16) ? 0 : 8;
#pragma unroll
    for (int nt = 0; nt < 8; nt++) {
      v8f accH = bias_splat(p.br2[f], nt, lane);
#pragma unroll
      for (int kc = 0; kc < 4; kc++)
        accH = wmma_f16(a_h[kc], load_btile(p.w2[f], 8, kc, nt, lane), accH);
      accH = silu8(accH);
      v8f accE = bias_splat(p.be2[f], nt, lane);
      accE = wmma_f16(a_e, load_btile(p.ew2[f], 8, 0, nt, lane), accE);
      accE = silu8(accE);
      const int col = nt * 16 + row;
      float* op = p.out[f] + (e0 + r0) * FDIM + col;
#pragma unroll
      for (int r = 0; r < 8; r++) op[(size_t)r * FDIM] = accH[r] + accE[r];
    }
  }
}

// ---------------------------------------------------------------------------
// Node update: t = [2*inv | ev_invs] @ w_int + b_int; out1 = 2*inv + t[:, :F];
// out2 = 2*ev * (1 + t[:, F + SEG]).  One wave per 16-node tile.
// ---------------------------------------------------------------------------
__global__ __launch_bounds__(256) void node_update_kernel(
    const float* __restrict__ invf, const float* __restrict__ evf,
    const _Float16* __restrict__ pk_wint, const float* __restrict__ bint,
    float* __restrict__ out1, float* __restrict__ out2) {
  __shared__ _Float16 sA[8 * 16 * 160];
  __shared__ float sBev[8 * 64];
  const int lane = threadIdx.x & 31, wid = threadIdx.x >> 5;
  const long tile = (long)blockIdx.x * 8 + wid;
  if (tile >= N_NODES / 16) return;  // wave-uniform early exit
  _Float16* stage = sA + wid * (16 * 160);
  float* bev = sBev + wid * 64;
  const int row = lane & 15;
  const long n0 = tile * 16;

  // Stage att_inv = 2*inv as f16 (cols 0..127); two lanes per row.
  {
    const int half = lane >> 4;
    const float* src = invf + (n0 + row) * FDIM + half * 64;
    _Float16* dst = stage + row * 160 + half * 64;
#pragma unroll
    for (int q = 0; q < 64; q++) dst[q] = (_Float16)(2.f * src[q]);
  }
  // ev invariants (cols 128..131) + zero pad (cols 132..159).
  if (lane < 16) {
    const float* es = evf + (n0 + row) * NCOMP;
    float i0 = 0.f, i1 = 0.f, i2 = 0.f, i3 = 0.f;
#pragma unroll
    for (int q = 0; q < NCOMP; q++) {
      const float a = es[q];
      const float sq = 4.f * a * a;  // (2*ev)^2
      if (q == 0) i0 += sq;
      else if (q < 4) i1 += sq;
      else if (q < 9) i2 += sq;
      else i3 += sq;
    }
    _Float16* dst = stage + row * 160 + 128;
    dst[0] = (_Float16)i0;
    dst[1] = (_Float16)i1;
    dst[2] = (_Float16)i2;
    dst[3] = (_Float16)i3;
#pragma unroll
    for (int q = 4; q < 16; q++) dst[q] = (_Float16)0.f;
  } else {
    _Float16* dst = stage + row * 160 + 144;
#pragma unroll
    for (int q = 0; q < 16; q++) dst[q] = (_Float16)0.f;
  }
  __builtin_amdgcn_wave_barrier();
  v16h A[5];
#pragma unroll
  for (int kc = 0; kc < 5; kc++) A[kc] = load_atile(stage, 160, lane, kc);
  __builtin_amdgcn_wave_barrier();

  const int r0 = (lane < 16) ? 0 : 8;
#pragma unroll
  for (int nt = 0; nt < 9; nt++) {
    const int col = nt * 16 + row;
    const float b = (col < 132) ? bint[col] : 0.f;
    v8f acc;
#pragma unroll
    for (int j = 0; j < 8; j++) acc[j] = b;
#pragma unroll
    for (int kc = 0; kc < 5; kc++)
      acc = wmma_f16(A[kc], load_btile(pk_wint, 9, kc, nt, lane), acc);
    if (nt < 8) {
#pragma unroll
      for (int r = 0; r < 8; r++) {
        const long node = n0 + r0 + r;
        out1[node * FDIM + col] = 2.f * invf[node * FDIM + col] + acc[r];
      }
    } else if (row < 4) {  // t[:, 128..131] -> LDS broadcast buffer
#pragma unroll
      for (int r = 0; r < 8; r++) bev[(r0 + r) * 4 + row] = acc[r];
    }
  }
  __builtin_amdgcn_wave_barrier();
  {
    const int rr = lane >> 1;
    const int c0 = (lane & 1) * 8;
    const long node = n0 + rr;
#pragma unroll
    for (int j = 0; j < 8; j++) {
      const int c = c0 + j;
      const int s = (c == 0) ? 0 : (c < 4) ? 1 : (c < 9) ? 2 : 3;
      const float att = 2.f * evf[node * NCOMP + c];
      out2[node * NCOMP + c] = att * (1.f + bev[rr * 4 + s]);
    }
  }
}

// ---------------------------------------------------------------------------
// Host launcher.
// ---------------------------------------------------------------------------
extern "C" void kernel_launch(void* const* d_in, const int* in_sizes, int n_in,
                              void* d_out, int out_size, void* d_ws,
                              size_t ws_size, hipStream_t stream) {
  (void)in_sizes; (void)n_in; (void)out_size; (void)ws_size;
  const float* invf = (const float*)d_in[0];
  const float* evf  = (const float*)d_in[1];
  const int* snd    = (const int*)d_in[2];
  const int* rcv    = (const int*)d_in[3];
  const float* len  = (const float*)d_in[5];
  const float* fw[2][8];
  for (int f = 0; f < 2; f++)
    for (int q = 0; q < 8; q++) fw[f][q] = (const float*)d_in[7 + f * 8 + q];
  // q: 0=rbf_w1 1=rbf_b1 2=rbf_w2 3=rbf_b2 4=ev_w1 5=ev_b1 6=ev_w2 7=ev_b2
  const float* wint = (const float*)d_in[23];
  const float* bint = (const float*)d_in[24];

  _Float16* ws = (_Float16*)d_ws;
  _Float16 *pk_w1[2], *pk_w2[2], *pk_ew1[2], *pk_ew2[2];
  size_t off = 0;
  for (int f = 0; f < 2; f++) {
    pk_w1[f]  = ws + off; off += 8  * 512;  // 32x128  -> 1x8 tiles
    pk_w2[f]  = ws + off; off += 32 * 512;  // 128x128 -> 4x8 tiles
    pk_ew1[f] = ws + off; off += 2  * 512;  // 4x32    -> 1x2 tiles
    pk_ew2[f] = ws + off; off += 8  * 512;  // 32x128  -> 1x8 tiles
  }
  _Float16* pk_wint = ws + off; off += 45 * 512;  // 132x132 -> 5x9 tiles

  for (int f = 0; f < 2; f++) {
    pack_weight_kernel<<<8, 32, 0, stream>>>(fw[f][0], pk_w1[f], 32, 128);
    pack_weight_kernel<<<32, 32, 0, stream>>>(fw[f][2], pk_w2[f], 128, 128);
    pack_weight_kernel<<<2, 32, 0, stream>>>(fw[f][4], pk_ew1[f], 4, 32);
    pack_weight_kernel<<<8, 32, 0, stream>>>(fw[f][6], pk_ew2[f], 32, 128);
  }
  pack_weight_kernel<<<45, 32, 0, stream>>>(wint, pk_wint, 132, 132);

  float* out1 = (float*)d_out;
  float* out2 = out1 + (size_t)N_NODES * FDIM;
  float* fwi  = out2 + (size_t)N_NODES * NCOMP;
  float* fwe  = fwi + (size_t)N_EDGES * FDIM;

  node_update_kernel<<<(N_NODES / 16 + 7) / 8, 256, 0, stream>>>(
      invf, evf, pk_wint, bint, out1, out2);

  EdgeParams ep;
  ep.lengths = len; ep.senders = snd; ep.receivers = rcv; ep.ev = evf;
  for (int f = 0; f < 2; f++) {
    ep.w1[f] = pk_w1[f]; ep.w2[f] = pk_w2[f];
    ep.ew1[f] = pk_ew1[f]; ep.ew2[f] = pk_ew2[f];
    ep.br1[f] = fw[f][1]; ep.br2[f] = fw[f][3];
    ep.be1[f] = fw[f][5]; ep.be2[f] = fw[f][7];
  }
  ep.out[0] = fwi; ep.out[1] = fwe;
  edge_filter_kernel<<<N_EDGES / (16 * 8), 256, 0, stream>>>(ep);
}
